// QwenMoe_78331613545164
// MI455X (gfx1250) — matrix operand order repaired and checked by reading, not compiled, not run on hardware
//
#include <hip/hip_runtime.h>
#include <cstdint>

// Problem sizes (fixed by the reference)
#define TT  1024   // tokens
#define HH  2048   // hidden
#define II  1408   // expert intermediate
#define EE  60     // routed experts
#define SSI 5632   // shared intermediate
#define TOPK 4
// h scratch: routed assignments occupy rows [0, 4*TT) of length II,
// shared-expert rows (length SSI) start at this element offset.
#define SHARED_HOFF (4 * TT * II)

typedef __attribute__((ext_vector_type(16))) __bf16        v16bf;
typedef __attribute__((ext_vector_type(16))) float         v16f;
typedef __attribute__((ext_vector_type(8)))  float         v8f;
typedef __attribute__((ext_vector_type(4)))  unsigned int  uint32x4;
typedef __attribute__((ext_vector_type(8)))  int           int32x8;
typedef __attribute__((ext_vector_type(4)))  int           int32x4;

union BF16x16 { v16bf v; unsigned int u[8]; };

__device__ __forceinline__ unsigned short f2bf(float f) {
  unsigned int x = __float_as_uint(f);
  return (unsigned short)((x + 0x7FFFu + ((x >> 16) & 1u)) >> 16);
}

// Load 16 consecutive fp32 from LDS (16B aligned) and convert to bf16x16
// using the native bf16 converters (v_cvt_*bf16_f32) via convertvector.
__device__ __forceinline__ v16bf cvt_row16(const float* p) {
  const float4* q = (const float4*)p;
  float4 a0 = q[0], a1 = q[1], a2 = q[2], a3 = q[3];
  v16f f;
  f[0] = a0.x;  f[1] = a0.y;  f[2] = a0.z;  f[3] = a0.w;
  f[4] = a1.x;  f[5] = a1.y;  f[6] = a1.z;  f[7] = a1.w;
  f[8] = a2.x;  f[9] = a2.y;  f[10] = a2.z; f[11] = a2.w;
  f[12] = a3.x; f[13] = a3.y; f[14] = a3.z; f[15] = a3.w;
  return __builtin_convertvector(f, v16bf);
}

// A operand (contiguous k order, same as B): lanes 0-15 hold K{s..s+15} with s=0,
// lanes 16-31 with s=16. Two 16B LDS reads per lane.
__device__ __forceinline__ v16bf load_a(const unsigned short* shA, unsigned lane) {
  int m = (int)(lane & 15);
  int s = (lane >> 4) ? 16 : 0;
  uint4 q0 = *(const uint4*)&shA[m * 32 + s];
  uint4 q1 = *(const uint4*)&shA[m * 32 + s + 8];
  BF16x16 a;
  a.u[0] = q0.x; a.u[1] = q0.y; a.u[2] = q0.z; a.u[3] = q0.w;
  a.u[4] = q1.x; a.u[5] = q1.y; a.u[6] = q1.z; a.u[7] = q1.w;
  return a.v;
}

// ---------------------------------------------------------------------------
// Tensor Data Mover: async-load a 2D fp32 tile (32 cols x `rows` rows, row
// stride `stride_elems`) from global memory into LDS at `lds_dst`.
// Descriptor packing per CDNA5 ISA. Tracked by TENSORcnt.
// ---------------------------------------------------------------------------
__device__ __forceinline__ void tdm_load_tile32(void* lds_dst, const float* gsrc,
                                                unsigned rows, unsigned stride_elems) {
  unsigned long long ga = (unsigned long long)(uintptr_t)gsrc;
  unsigned lds_off = (unsigned)(uintptr_t)lds_dst;   // low 32 bits == LDS byte offset
  uint32x4 g0;
  g0.x = 1u;                                          // count=1, user descriptor
  g0.y = lds_off;                                     // lds_addr [63:32]
  g0.z = (unsigned)(ga & 0xFFFFFFFFull);              // global_addr [95:64]
  g0.w = (unsigned)((ga >> 32) & 0x1FFFFFFull)        // global_addr [120:96]
       | (2u << 30);                                  // type=2 ("image")
  int32x8 g1;
  g1[0] = (int)(2u << 16);                            // data_size=2 (4B)
  g1[1] = (int)(32u << 16);                           // tensor_dim0 = 32
  g1[2] = (int)(rows << 16);                          // tensor_dim1 lo16 = rows
  g1[3] = (int)(((rows >> 16) & 0xFFFFu) | (32u << 16)); // tensor_dim1 hi | tile_dim0=32
  g1[4] = (int)(rows & 0xFFFFu);                      // tile_dim1 = rows, tile_dim2=0
  g1[5] = (int)stride_elems;                          // tensor_dim0_stride lo32
  g1[6] = 0;
  g1[7] = 0;
  int32x4 z4 = {0, 0, 0, 0};
#if defined(__clang_major__) && (__clang_major__ >= 23)
  int32x8 z8 = {0, 0, 0, 0, 0, 0, 0, 0};
  __builtin_amdgcn_tensor_load_to_lds(g0, g1, z4, z4, z8, 0);
#else
  __builtin_amdgcn_tensor_load_to_lds(g0, g1, z4, z4, 0);
#endif
}

// ---------------------------------------------------------------------------
// Kernel 1: router (1 block/token, 64 threads). Stages x row in LDS, emits
// bf16 copy of x, 61 dot products (60 router + shared gate), softmax + top-4,
// builds per-expert compacted lists via atomics.
// ---------------------------------------------------------------------------
__global__ __launch_bounds__(64, 1) void moe_router(
    const float* __restrict__ x, const float* __restrict__ router_w,
    const float* __restrict__ shared_gate_w,
    unsigned short* __restrict__ xbf, int* __restrict__ cnt,
    int* __restrict__ tok_list, int* __restrict__ hoff_list,
    float* __restrict__ scale_list) {
  const int t = blockIdx.x;
  __shared__ float shX[HH];
  __shared__ float shL[64];
  const int tid = threadIdx.x;

  const float4* xr = (const float4*)(x + (size_t)t * HH);
  for (int idx = tid; idx < HH / 4; idx += 64) {
    float4 v = xr[idx];
    ((float4*)shX)[idx] = v;
    uint2 p;
    p.x = (unsigned)f2bf(v.x) | ((unsigned)f2bf(v.y) << 16);
    p.y = (unsigned)f2bf(v.z) | ((unsigned)f2bf(v.w) << 16);
    ((uint2*)(xbf + (size_t)t * HH))[idx] = p;
  }
  __syncthreads();

  if (tid <= EE) {
    const float* w = (tid == EE) ? shared_gate_w : (router_w + (size_t)tid * HH);
    float s = 0.f;
    for (int h = 0; h < HH; ++h) s = fmaf(shX[h], w[h], s);
    shL[tid] = s;
  }
  __syncthreads();

  if (tid == 0) {
    float mx = -3.0e38f;
    for (int e = 0; e < EE; ++e) mx = fmaxf(mx, shL[e]);
    float sum = 0.f;
    for (int e = 0; e < EE; ++e) { float p = __expf(shL[e] - mx); shL[e] = p; sum += p; }
    const float inv = __builtin_amdgcn_rcpf(sum);
    for (int j = 0; j < TOPK; ++j) {
      int be = 0; float bp = -1.f;
      for (int e = 0; e < EE; ++e)
        if (shL[e] > bp) { bp = shL[e]; be = e; }
      shL[be] = -2.f;
      int pos = atomicAdd(&cnt[be], 1);
      tok_list[be * TT + pos]   = t;
      hoff_list[be * TT + pos]  = (t * TOPK + j) * II;
      scale_list[be * TT + pos] = bp * inv;
    }
    float gl = shL[EE];
    scale_list[EE * TT + t] = __builtin_amdgcn_rcpf(1.f + __expf(-gl));
  }
}

// ---------------------------------------------------------------------------
// Kernel 2: gate/up GEMMs + SiLU, h = silu(x Wg^T) * (x Wu^T), stored bf16.
// grid = (SSI/64 i-tiles, 61 experts); block = 128 threads = 4 waves.
// Macro-tile: M=16 gathered tokens x N=64 (16 per wave), K-loop of 32.
// Weight tiles stream through DOUBLE-BUFFERED LDS via the TDM so the
// tensor DMA overlaps the WMMAs (one barrier + one s_wait_tensorcnt / iter).
// ---------------------------------------------------------------------------
__global__ __launch_bounds__(128, 1) void moe_gateup(
    const float* __restrict__ w_gate, const float* __restrict__ w_up,
    const float* __restrict__ sw_gate, const float* __restrict__ sw_up,
    const unsigned short* __restrict__ xbf,
    const int* __restrict__ cnt, const int* __restrict__ tok_list,
    const int* __restrict__ hoff_list, unsigned short* __restrict__ h_all) {
  const int e = blockIdx.y;
  const int nI = (e == EE) ? SSI : II;
  const int iTiles = nI >> 6;
  if ((int)blockIdx.x >= iTiles) return;
  const int count = (e == EE) ? TT : cnt[e];
  if (count == 0) return;
  const float* Wg = (e == EE) ? sw_gate : (w_gate + (size_t)e * II * HH);
  const float* Wu = (e == EE) ? sw_up   : (w_up   + (size_t)e * II * HH);
  const int i_base = (int)blockIdx.x << 6;

  __shared__ float shBG[2][64 * 32];        // 2 x 8 KB gate tile (fp32 via TDM)
  __shared__ float shBU[2][64 * 32];        // 2 x 8 KB up tile
  __shared__ unsigned short shA[2][16 * 32];// 2 x 1 KB bf16 x tile (gathered)
  __shared__ int shTok[16];
  __shared__ int shHOff[16];

  const unsigned tid  = threadIdx.x;
  const unsigned wave = tid >> 5;
  const unsigned lane = tid & 31;
  const unsigned hi   = lane >> 4;
  const unsigned nrow = wave * 16 + (lane & 15);
  const int mTiles = (count + 15) >> 4;
  const int NS = HH / 32;

  for (int mt = 0; mt < mTiles; ++mt) {
    __syncthreads();
    if (tid < 16) {
      int pos = mt * 16 + (int)tid;
      int tok = 0, hoff = -1;
      if (pos < count) {
        if (e == EE) { tok = pos; hoff = SHARED_HOFF + pos * SSI; }
        else         { tok = tok_list[e * TT + pos]; hoff = hoff_list[e * TT + pos]; }
      }
      shTok[tid] = tok; shHOff[tid] = hoff;
    }
    __syncthreads();

    v8f accG, accU;
#pragma unroll
    for (int q = 0; q < 8; ++q) { accG[q] = 0.f; accU[q] = 0.f; }

    // ---- pipeline prologue: fill stage 0 ----
    if (wave == 0) {
      tdm_load_tile32(&shBG[0][0], Wg + (size_t)i_base * HH, 64, HH);
      tdm_load_tile32(&shBU[0][0], Wu + (size_t)i_base * HH, 64, HH);
    }
    if (tid >= 64) {                         // waves 2,3 stage A tile 0
      int j = (int)tid - 64, m = j >> 2, c = j & 3;
      *(uint4*)&shA[0][m * 32 + c * 8] =
          *(const uint4*)(xbf + (size_t)shTok[m] * HH + c * 8);
    }
    if (wave == 0) __builtin_amdgcn_s_wait_tensorcnt(0);
    __syncthreads();

    // ---- steady state: prefetch s+1 while computing s ----
    for (int s = 0; s < NS; ++s) {
      const int p  = s & 1;
      const int pn = p ^ 1;
      const int k0n = (s + 1) * 32;
      if (s + 1 < NS) {
        if (wave == 0) {
          tdm_load_tile32(&shBG[pn][0], Wg + (size_t)i_base * HH + k0n, 64, HH);
          tdm_load_tile32(&shBU[pn][0], Wu + (size_t)i_base * HH + k0n, 64, HH);
        }
        if (tid >= 64) {
          int j = (int)tid - 64, m = j >> 2, c = j & 3;
          *(uint4*)&shA[pn][m * 32 + c * 8] =
              *(const uint4*)(xbf + (size_t)shTok[m] * HH + k0n + c * 8);
        }
      }
      // compute stage s
      v16bf a  = load_a(&shA[p][0], lane);
      v16bf bg = cvt_row16(&shBG[p][nrow * 32 + hi * 16]);
      v16bf bu = cvt_row16(&shBU[p][nrow * 32 + hi * 16]);
      accG = __builtin_amdgcn_wmma_f32_16x16x32_bf16(false, a, false, bg,
                                                     (short)0, accG, false, false);
      accU = __builtin_amdgcn_wmma_f32_16x16x32_bf16(false, a, false, bu,
                                                     (short)0, accU, false, false);
      if (wave == 0) __builtin_amdgcn_s_wait_tensorcnt(0);  // next tiles landed
      __syncthreads();
    }

    // D layout: M = r + 8*hi, N = lane%16. SiLU(g)*u -> bf16 h row.
    const int nCol = i_base + (int)nrow;
    __bf16* h16 = (__bf16*)h_all;
#pragma unroll
    for (int r = 0; r < 8; ++r) {
      int m = r + (hi ? 8 : 0);
      int hoff = shHOff[m];
      if (hoff >= 0) {
        float g = accG[r], u = accU[r];
        float hv = g * __builtin_amdgcn_rcpf(1.f + __expf(-g)) * u;
        h16[(size_t)hoff + nCol] = (__bf16)hv;
      }
    }
  }
}

// ---------------------------------------------------------------------------
// Kernel 3: down projection. out[t,:] += scale * (h @ Wd^T) via f32 atomics.
// grid = (HH/128 n-tiles, 61 experts); block = 256 threads = 8 waves.
// Same double-buffered TDM pipeline, K = I_e.
// ---------------------------------------------------------------------------
__global__ __launch_bounds__(256, 1) void moe_down(
    const float* __restrict__ w_down, const float* __restrict__ sw_down,
    const unsigned short* __restrict__ h_all,
    const int* __restrict__ cnt, const int* __restrict__ tok_list,
    const int* __restrict__ hoff_list, const float* __restrict__ scale_list,
    float* __restrict__ out) {
  const int e = blockIdx.y;
  const int nK = (e == EE) ? SSI : II;
  const int count = (e == EE) ? TT : cnt[e];
  if (count == 0) return;
  const float* Wd = (e == EE) ? sw_down : (w_down + (size_t)e * HH * II);
  const int n_base = (int)blockIdx.x << 7;

  __shared__ float shB[2][128 * 32];        // 2 x 16 KB w_down tile (fp32 via TDM)
  __shared__ unsigned short shA[2][16 * 32];
  __shared__ int shTok[16];
  __shared__ int shHOff[16];
  __shared__ float shSc[16];

  const unsigned tid  = threadIdx.x;
  const unsigned wave = tid >> 5;
  const unsigned lane = tid & 31;
  const unsigned hi   = lane >> 4;
  const unsigned nrow = wave * 16 + (lane & 15);
  const int mTiles = (count + 15) >> 4;
  const int NS = nK / 32;

  for (int mt = 0; mt < mTiles; ++mt) {
    __syncthreads();
    if (tid < 16) {
      int pos = mt * 16 + (int)tid;
      int tok = 0, hoff = 0; float sc = 0.f;     // padded rows: scale 0 -> harmless
      if (pos < count) {
        if (e == EE) { tok = pos; hoff = SHARED_HOFF + pos * SSI; sc = scale_list[EE * TT + pos]; }
        else {
          tok  = tok_list[e * TT + pos];
          hoff = hoff_list[e * TT + pos];
          sc   = scale_list[e * TT + pos];
        }
      }
      shTok[tid] = tok; shHOff[tid] = hoff; shSc[tid] = sc;
    }
    __syncthreads();

    v8f acc;
#pragma unroll
    for (int q = 0; q < 8; ++q) acc[q] = 0.f;

    // prologue
    if (wave == 0)
      tdm_load_tile32(&shB[0][0], Wd + (size_t)n_base * nK, 128, (unsigned)nK);
    if (tid >= 64 && tid < 128) {
      int j = (int)tid - 64, m = j >> 2, c = j & 3;
      *(uint4*)&shA[0][m * 32 + c * 8] =
          *(const uint4*)(h_all + (size_t)shHOff[m] + c * 8);
    }
    if (wave == 0) __builtin_amdgcn_s_wait_tensorcnt(0);
    __syncthreads();

    for (int s = 0; s < NS; ++s) {
      const int p  = s & 1;
      const int pn = p ^ 1;
      const int k0n = (s + 1) * 32;
      if (s + 1 < NS) {
        if (wave == 0)
          tdm_load_tile32(&shB[pn][0], Wd + (size_t)n_base * nK + k0n, 128, (unsigned)nK);
        if (tid >= 64 && tid < 128) {
          int j = (int)tid - 64, m = j >> 2, c = j & 3;
          *(uint4*)&shA[pn][m * 32 + c * 8] =
              *(const uint4*)(h_all + (size_t)shHOff[m] + k0n + c * 8);
        }
      }
      v16bf a = load_a(&shA[p][0], lane);
      v16bf b = cvt_row16(&shB[p][nrow * 32 + hi * 16]);
      acc = __builtin_amdgcn_wmma_f32_16x16x32_bf16(false, a, false, b,
                                                    (short)0, acc, false, false);
      if (wave == 0) __builtin_amdgcn_s_wait_tensorcnt(0);
      __syncthreads();
    }

    const int n = n_base + (int)nrow;
#pragma unroll
    for (int r = 0; r < 8; ++r) {
      int m = r + (hi ? 8 : 0);
      atomicAdd(&out[(size_t)shTok[m] * HH + n], shSc[m] * acc[r]);
    }
  }
}

// ---------------------------------------------------------------------------
// Launch: memsets + 3 kernels, all on `stream` (graph-capture safe).
// Workspace (~28 MB): xbf16 | counts | tok_list | hoff_list | scale_list |
// h scratch (routed rows, then shared rows), 256B aligned.
// ---------------------------------------------------------------------------
extern "C" void kernel_launch(void* const* d_in, const int* in_sizes, int n_in,
                              void* d_out, int out_size, void* d_ws, size_t ws_size,
                              hipStream_t stream) {
  const float* x             = (const float*)d_in[0];
  const float* router_w      = (const float*)d_in[1];
  const float* w_gate        = (const float*)d_in[2];
  const float* w_up          = (const float*)d_in[3];
  const float* w_down        = (const float*)d_in[4];
  const float* sw_gate       = (const float*)d_in[5];
  const float* sw_up         = (const float*)d_in[6];
  const float* sw_down       = (const float*)d_in[7];
  const float* shared_gate_w = (const float*)d_in[8];
  float* out = (float*)d_out;
  (void)in_sizes; (void)n_in; (void)ws_size;

  char* ws = (char*)d_ws;
  size_t off = 0;
  auto carve = [&](size_t bytes) -> void* {
    void* p = ws + off;
    off = (off + bytes + 255) & ~(size_t)255;
    return p;
  };
  unsigned short* xbf   = (unsigned short*)carve((size_t)TT * HH * 2);
  int*   cnt            = (int*)carve(64 * sizeof(int));
  int*   tok_list       = (int*)carve((size_t)(EE + 1) * TT * sizeof(int));
  int*   hoff_list      = (int*)carve((size_t)(EE + 1) * TT * sizeof(int));
  float* scale_list     = (float*)carve((size_t)(EE + 1) * TT * sizeof(float));
  unsigned short* h_all = (unsigned short*)carve((size_t)(4 * TT * II + TT * SSI) * 2);

  hipMemsetAsync(cnt, 0, 64 * sizeof(int), stream);
  hipMemsetAsync(out, 0, (size_t)out_size * sizeof(float), stream);

  moe_router<<<TT, 64, 0, stream>>>(x, router_w, shared_gate_w, xbf, cnt,
                                    tok_list, hoff_list, scale_list);
  moe_gateup<<<dim3(SSI / 64, EE + 1), 128, 0, stream>>>(
      w_gate, w_up, sw_gate, sw_up, xbf, cnt, tok_list, hoff_list, h_all);
  moe_down<<<dim3(HH / 128, EE + 1), 256, 0, stream>>>(
      w_down, sw_down, h_all, cnt, tok_list, hoff_list, scale_list, out);
}
